// MultiHeadAttention_48833778155608
// MI455X (gfx1250) — matrix operand order repaired and checked by reading, not compiled, not run on hardware
//
#include <hip/hip_runtime.h>
#include <hip/hip_bf16.h>

#define HIDDEN   1024
#define HEADS    16
#define HEAD_DIM 64
#define SEQ      2048
#define BATCH    2

typedef __attribute__((ext_vector_type(16))) _Float16 v16h;
typedef __attribute__((ext_vector_type(8)))  _Float16 v8h;
typedef __attribute__((ext_vector_type(8)))  float    v8f;

__device__ __forceinline__ v8f wmma_f16f32(v16h a, v16h b, v8f c) {
  // D = A(16x32 f16) * B(32x16 f16) + C(16x16 f32)
  return __builtin_amdgcn_wmma_f32_16x16x32_f16(false, a, false, b, (short)0, c,
                                                false, false);
}

// ---------------------------------------------------------------------------
// Fragment loaders (per CDNA5 ISA 7.12.2 layouts)
// A (MxK = 16x32, 16-bit): lane m(0-15)/m+16; K in blocks of 8 per half-wave:
//   lanes 0-15 : halves 0-7 = K[k0..k0+7],    halves 8-15 = K[k0+16..k0+23]
//   lanes 16-31: halves 0-7 = K[k0+8..k0+15], halves 8-15 = K[k0+24..k0+31]
// Source is row-major [rows=M, cols=K].
// ---------------------------------------------------------------------------
__device__ __forceinline__ v16h loadA_f16(const _Float16* __restrict__ base, int stride,
                                          int row0, int k0, int lane) {
  const v8h* p = (const v8h*)(base + (size_t)(row0 + (lane & 15)) * stride +
                              k0 + (lane >> 4) * 8);
  v8h a = p[0], b = p[2];
  v16h f;
#pragma unroll
  for (int i = 0; i < 8; ++i) { f[i] = a[i]; f[i + 8] = b[i]; }
  return f;
}

// ---------------------------------------------------------------------------
// B (KxN = 32x16, 16-bit): lane n(0-15); K in blocks of 8 per half-wave (same as A):
//   lanes 0-15 : halves 0-7 = K[k0..k0+7],    halves 8-15 = K[k0+16..k0+23]
//   lanes 16-31: halves 0-7 = K[k0+8..k0+15], halves 8-15 = K[k0+24..k0+31]
// Source is B^T row-major [rows=N, cols=K]  (torch-Linear weight layout).
// ---------------------------------------------------------------------------
__device__ __forceinline__ v16h loadB_f16(const _Float16* __restrict__ base, int stride,
                                          int row0, int k0, int lane) {
  const v8h* p = (const v8h*)(base + (size_t)(row0 + (lane & 15)) * stride +
                              k0 + (lane >> 4) * 8);
  v8h a = p[0], b = p[2];
  v16h f;
#pragma unroll
  for (int i = 0; i < 8; ++i) { f[i] = a[i]; f[i + 8] = b[i]; }
  return f;
}

// ---------------------------------------------------------------------------
// One-pass f32 -> f16 conversion (activations and weights), 8 elems/thread.
// ---------------------------------------------------------------------------
__global__ __launch_bounds__(256) void cvt_f16_kernel(const float* __restrict__ src,
                                                      _Float16* __restrict__ dst,
                                                      int n) {
  int i = (blockIdx.x * 256 + threadIdx.x) * 8;
  if (i >= n) return;
  const float4* p = (const float4*)(src + i);
  float4 a = p[0], b = p[1];
  v8h o;
  o[0] = (_Float16)a.x; o[1] = (_Float16)a.y; o[2] = (_Float16)a.z; o[3] = (_Float16)a.w;
  o[4] = (_Float16)b.x; o[5] = (_Float16)b.y; o[6] = (_Float16)b.z; o[7] = (_Float16)b.w;
  *(v8h*)(dst + i) = o;
}

// ---------------------------------------------------------------------------
// Projection GEMM (all-f16 operands): O[m,n] = X[m,:] . W[n,:] + bias[n]
// OMODE 0: store f16 as [B,H,S,D]   (Q, K)
// OMODE 1: store f16 as [B,H,D,S]   (V transposed, for the PV B-fragments)
// block = 128 threads (4 waves, 2x2); wave tile 32x64 -> 8 WMMA per K-step.
// ---------------------------------------------------------------------------
template <int OMODE>
__global__ __launch_bounds__(128) void proj_kernel(const _Float16* __restrict__ X,
                                                   const _Float16* __restrict__ W,
                                                   const float* __restrict__ bias,
                                                   _Float16* __restrict__ outh) {
  const int lane = threadIdx.x & 31;
  const int wave = threadIdx.x >> 5;
  const int m0 = blockIdx.y * 64 + (wave >> 1) * 32;
  const int n0 = blockIdx.x * 128 + (wave & 1) * 64;

  v8f z = {};
  v8f acc[2][4] = {{z, z, z, z}, {z, z, z, z}};

  for (int k = 0; k < HIDDEN; k += 32) {
    if (k + 32 < HIDDEN) {   // WGP-scope streaming prefetch of next K-step rows
      __builtin_prefetch(X + (size_t)(m0 + lane) * HIDDEN + k + 32, 0, 3);
      __builtin_prefetch(W + (size_t)(n0 + lane) * HIDDEN + k + 32, 0, 3);
      __builtin_prefetch(W + (size_t)(n0 + 32 + lane) * HIDDEN + k + 32, 0, 3);
    }
    v16h a0 = loadA_f16(X, HIDDEN, m0,      k, lane);
    v16h a1 = loadA_f16(X, HIDDEN, m0 + 16, k, lane);
    v16h b[4];
#pragma unroll
    for (int j = 0; j < 4; ++j) b[j] = loadB_f16(W, HIDDEN, n0 + j * 16, k, lane);
#pragma unroll
    for (int j = 0; j < 4; ++j) {
      acc[0][j] = wmma_f16f32(a0, b[j], acc[0][j]);
      acc[1][j] = wmma_f16f32(a1, b[j], acc[1][j]);
    }
  }

  const int col = lane & 15, hi = lane >> 4;
#pragma unroll
  for (int i = 0; i < 2; ++i)
#pragma unroll
    for (int j = 0; j < 4; ++j) {
      int n = n0 + j * 16 + col;
      float bv = bias[n];
      int h = n >> 6, d = n & 63;
#pragma unroll
      for (int v = 0; v < 8; ++v) {
        int m = m0 + i * 16 + hi * 8 + v;
        int b2 = m >> 11, s = m & (SEQ - 1);
        float val = acc[i][j][v] + bv;
        if (OMODE == 0)
          outh[(((size_t)(b2 * HEADS + h)) * SEQ + s) * HEAD_DIM + d] = (_Float16)val;
        else
          outh[(((size_t)(b2 * HEADS + h)) * HEAD_DIM + d) * SEQ + s] = (_Float16)val;
      }
    }
}

// ---------------------------------------------------------------------------
// Output projection (f16 x f16 -> f32): out[m,n] = A[m,:] . W[n,:] + bias[n]
// ---------------------------------------------------------------------------
__global__ __launch_bounds__(128) void oproj_kernel(const _Float16* __restrict__ A,
                                                    const _Float16* __restrict__ W,
                                                    const float* __restrict__ bias,
                                                    float* __restrict__ out) {
  const int lane = threadIdx.x & 31;
  const int wave = threadIdx.x >> 5;
  const int m0 = blockIdx.y * 64 + (wave >> 1) * 32;
  const int n0 = blockIdx.x * 128 + (wave & 1) * 64;

  v8f z = {};
  v8f acc[2][4] = {{z, z, z, z}, {z, z, z, z}};

  for (int k = 0; k < HIDDEN; k += 32) {
    if (k + 32 < HIDDEN) {
      __builtin_prefetch(A + (size_t)(m0 + lane) * HIDDEN + k + 32, 0, 3);
      __builtin_prefetch(W + (size_t)(n0 + lane) * HIDDEN + k + 32, 0, 3);
      __builtin_prefetch(W + (size_t)(n0 + 32 + lane) * HIDDEN + k + 32, 0, 3);
    }
    v16h a0 = loadA_f16(A, HIDDEN, m0,      k, lane);
    v16h a1 = loadA_f16(A, HIDDEN, m0 + 16, k, lane);
    v16h b[4];
#pragma unroll
    for (int j = 0; j < 4; ++j) b[j] = loadB_f16(W, HIDDEN, n0 + j * 16, k, lane);
#pragma unroll
    for (int j = 0; j < 4; ++j) {
      acc[0][j] = wmma_f16f32(a0, b[j], acc[0][j]);
      acc[1][j] = wmma_f16f32(a1, b[j], acc[1][j]);
    }
  }

  const int col = lane & 15, hi = lane >> 4;
#pragma unroll
  for (int i = 0; i < 2; ++i)
#pragma unroll
    for (int j = 0; j < 4; ++j) {
      int n = n0 + j * 16 + col;
      float bv = bias[n];
#pragma unroll
      for (int v = 0; v < 8; ++v) {
        int m = m0 + i * 16 + hi * 8 + v;
        out[(size_t)m * HIDDEN + n] = acc[i][j][v] + bv;
      }
    }
}

// ---------------------------------------------------------------------------
// Flash attention (causal), one wave per 16-query tile, 8 waves per block.
// Per 32-key step: 4 WMMA (S = Q K^T, D=64) + 4 WMMA (P V).
// P staged through 1 KB LDS per wave to convert C-layout -> A-layout.
// ---------------------------------------------------------------------------
__global__ __launch_bounds__(256) void attn_kernel(const _Float16* __restrict__ Qh,
                                                   const _Float16* __restrict__ Kh,
                                                   const _Float16* __restrict__ Vt,
                                                   _Float16* __restrict__ Oh) {
  __shared__ _Float16 lds_p[8][16 * 32];

  const int lane = threadIdx.x & 31;
  const int wave = threadIdx.x >> 5;
  const int bh = blockIdx.y;                       // b*HEADS + h
  const int q0 = (blockIdx.x * 8 + wave) * 16;
  const int col = lane & 15, hi = lane >> 4;

  const _Float16* Qb = Qh + (size_t)bh * SEQ * HEAD_DIM;
  const _Float16* Kb = Kh + (size_t)bh * SEQ * HEAD_DIM;
  const _Float16* Vb = Vt + (size_t)bh * HEAD_DIM * SEQ;

  v16h aq0 = loadA_f16(Qb, HEAD_DIM, q0, 0, lane);
  v16h aq1 = loadA_f16(Qb, HEAD_DIM, q0, 32, lane);

  v8f z = {};
  v8f acc[4] = {z, z, z, z};
  float m_i[8], l_i[8];
#pragma unroll
  for (int v = 0; v < 8; ++v) { m_i[v] = -1e30f; l_i[v] = 0.0f; }

  _Float16* pbuf = lds_p[wave];
  const float scale = 0.125f;                      // 1/sqrt(64)

  for (int j = 0; j <= q0 + 15; j += 32) {
    if (j + 32 <= q0 + 15) {                       // prefetch next key tile
      __builtin_prefetch(Kb + (size_t)(j + 32 + lane) * HEAD_DIM, 0, 3);
      __builtin_prefetch(Vb + (size_t)lane * SEQ + j + 32, 0, 3);
      __builtin_prefetch(Vb + (size_t)(32 + lane) * SEQ + j + 32, 0, 3);
    }

    // ---- scores: two 16x16 tiles covering keys [j, j+32) ----
    v8f s0 = {}, s1 = {};
    {
      v16h b0 = loadB_f16(Kb, HEAD_DIM, j,      0,  lane);
      v16h b1 = loadB_f16(Kb, HEAD_DIM, j,      32, lane);
      s0 = wmma_f16f32(aq0, b0, s0);
      s0 = wmma_f16f32(aq1, b1, s0);
      v16h b2 = loadB_f16(Kb, HEAD_DIM, j + 16, 0,  lane);
      v16h b3 = loadB_f16(Kb, HEAD_DIM, j + 16, 32, lane);
      s1 = wmma_f16f32(aq0, b2, s1);
      s1 = wmma_f16f32(aq1, b3, s1);
    }

    // ---- causal mask + online softmax update ----
#pragma unroll
    for (int v = 0; v < 8; ++v) {
      int q = q0 + hi * 8 + v;
      float r0 = s0[v] * scale;
      float r1 = s1[v] * scale;
      if (j + col > q)      r0 = -1e30f;
      if (j + 16 + col > q) r1 = -1e30f;
      float rmax = fmaxf(r0, r1);
#pragma unroll
      for (int off = 1; off < 16; off <<= 1)
        rmax = fmaxf(rmax, __shfl_xor(rmax, off, 32));
      float mn = fmaxf(m_i[v], rmax);
      float p0 = __expf(r0 - mn);
      float p1 = __expf(r1 - mn);
      float rs = p0 + p1;
#pragma unroll
      for (int off = 1; off < 16; off <<= 1)
        rs += __shfl_xor(rs, off, 32);
      float corr = __expf(m_i[v] - mn);
      l_i[v] = l_i[v] * corr + rs;
      m_i[v] = mn;
#pragma unroll
      for (int f = 0; f < 4; ++f) acc[f][v] *= corr;
      s0[v] = p0;
      s1[v] = p1;
    }

    // ---- stage P (16x32) to LDS, reload as A-fragment ----
#pragma unroll
    for (int v = 0; v < 8; ++v) {
      int row = hi * 8 + v;
      pbuf[row * 32 + col]      = (_Float16)s0[v];
      pbuf[row * 32 + 16 + col] = (_Float16)s1[v];
    }
    v16h ap = loadA_f16(pbuf, 32, 0, 0, lane);

    // ---- O += P * V ----
#pragma unroll
    for (int f = 0; f < 4; ++f) {
      v16h bvf = loadB_f16(Vb, SEQ, f * 16, j, lane);
      acc[f] = wmma_f16f32(ap, bvf, acc[f]);
    }
  }

  // ---- normalize and write [B*S, HIDDEN] rows of f16 ----
  const int b = bh >> 4, h = bh & 15;
#pragma unroll
  for (int f = 0; f < 4; ++f)
#pragma unroll
    for (int v = 0; v < 8; ++v) {
      int q = q0 + hi * 8 + v;
      float o = acc[f][v] / l_i[v];
      Oh[((size_t)(b * SEQ + q)) * HIDDEN + h * HEAD_DIM + f * 16 + col] = (_Float16)o;
    }
}

// ---------------------------------------------------------------------------
extern "C" void kernel_launch(void* const* d_in, const int* in_sizes, int n_in,
                              void* d_out, int out_size, void* d_ws, size_t ws_size,
                              hipStream_t stream) {
  const float* q  = (const float*)d_in[0];
  const float* k  = (const float*)d_in[1];
  const float* v  = (const float*)d_in[2];
  const float* Wq = (const float*)d_in[3];
  const float* bq = (const float*)d_in[4];
  const float* Wk = (const float*)d_in[5];
  const float* bk = (const float*)d_in[6];
  const float* Wv = (const float*)d_in[7];
  const float* bv = (const float*)d_in[8];
  const float* Wo = (const float*)d_in[9];
  const float* bo = (const float*)d_in[10];

  // workspace layout (halfs):
  //   Xq|Xk|Xv (f16 activations, 4M each) | Wq|Wk|Wv|Wo (f16 weights, 1M each)
  //   Qh|Kh|Vt|Ah (4M each)                             -> 32M halfs = 64 MB
  const size_t nTok = (size_t)BATCH * SEQ * HIDDEN;     // 4M
  const size_t nW   = (size_t)HIDDEN * HIDDEN;          // 1M
  _Float16* Xq  = (_Float16*)d_ws;
  _Float16* Xk  = Xq + nTok;
  _Float16* Xv  = Xk + nTok;
  _Float16* Wqh = Xv + nTok;
  _Float16* Wkh = Wqh + nW;
  _Float16* Wvh = Wkh + nW;
  _Float16* Woh = Wvh + nW;
  _Float16* Qh  = Woh + nW;
  _Float16* Kh  = Qh + nTok;
  _Float16* Vt  = Kh + nTok;
  _Float16* Ah  = Vt + nTok;

  // f32 -> f16 conversion passes
  {
    int nt = (int)nTok, nw = (int)nW;
    int gt = (nt / 8 + 255) / 256, gw = (nw / 8 + 255) / 256;
    cvt_f16_kernel<<<gt, 256, 0, stream>>>(q, Xq, nt);
    cvt_f16_kernel<<<gt, 256, 0, stream>>>(k, Xk, nt);
    cvt_f16_kernel<<<gt, 256, 0, stream>>>(v, Xv, nt);
    cvt_f16_kernel<<<gw, 256, 0, stream>>>(Wq, Wqh, nw);
    cvt_f16_kernel<<<gw, 256, 0, stream>>>(Wk, Wkh, nw);
    cvt_f16_kernel<<<gw, 256, 0, stream>>>(Wv, Wvh, nw);
    cvt_f16_kernel<<<gw, 256, 0, stream>>>(Wo, Woh, nw);
  }

  dim3 gp(HIDDEN / 128, (BATCH * SEQ) / 64);  // 8 x 64 blocks
  dim3 bp(128);
  proj_kernel<0><<<gp, bp, 0, stream>>>(Xq, Wqh, bq, Qh);
  proj_kernel<0><<<gp, bp, 0, stream>>>(Xk, Wkh, bk, Kh);
  proj_kernel<1><<<gp, bp, 0, stream>>>(Xv, Wvh, bv, Vt);

  dim3 ga(SEQ / 128, BATCH * HEADS);          // 16 x 32 blocks, 8 waves each
  attn_kernel<<<ga, dim3(256), 0, stream>>>(Qh, Kh, Vt, Ah);

  oproj_kernel<<<gp, bp, 0, stream>>>(Ah, Woh, bo, (float*)d_out);
}
